// myLSTM_88888643158230
// MI455X (gfx1250) — matrix operand order repaired and checked by reading, not compiled, not run on hardware
//
#include <hip/hip_runtime.h>
#include <hip/hip_bf16.h>
#include <math.h>

// ---------------------------------------------------------------------------
// LSTM forward for MI455X (gfx1250), wave32 + WMMA bf16.
//
// gates = [x_t ; h_prev] @ [Wx | Wh]^T + bx   (K = D + H = 2048)
// computed with v_wmma_f32_16x16x32_bf16, fp32 accumulation.
//
// Persistent kernel: 64 blocks x 512 threads (16 waves). Block b owns columns
// [16b, 16b+16) of H for ALL four gates, so the pointwise LSTM update is block
// local. Grid-wide sync between timesteps via a global atomic barrier.
// x_t is converted f32->bf16 into LDS; h_prev (already bf16 in ws) is staged
// with GLOBAL_LOAD_ASYNC_TO_LDS_B128 (ASYNCcnt). Weights live in L2 as bf16.
// Pointwise uses hardware v_tanh_f32 / v_exp_f32 instead of libm.
// ---------------------------------------------------------------------------

typedef __attribute__((ext_vector_type(16))) __bf16 v16bf;
typedef __attribute__((ext_vector_type(8)))  __bf16 v8bf;
typedef __attribute__((ext_vector_type(8)))  float  v8f;
typedef __attribute__((ext_vector_type(4)))  float  v4f;   // clang vector: OK for nontemporal builtins

#define T_STEPS 512
#define BATCH   64
#define DDIM    1024
#define HDIM    1024
#define KDIM    2048   // D + H
#define GDIM    4096   // 4*H

// workspace layout (bytes)
#define WCAT_BYTES ((size_t)GDIM * KDIM * 2)            // 16 MiB bf16 [4096][2048]
#define H_BYTES    ((size_t)BATCH * HDIM * 2)           // 128 KiB bf16
#define C_BYTES    ((size_t)BATCH * HDIM * 4)           // 256 KiB f32
#define OFF_H0     (WCAT_BYTES)
#define OFF_H1     (OFF_H0 + H_BYTES)
#define OFF_C      (OFF_H1 + H_BYTES)
#define OFF_BAR    (OFF_C + C_BYTES)
#define WS_NEEDED  (OFF_BAR + 256)

#define SMEM_ACAT_BYTES ((size_t)BATCH * KDIM * 2)          // 256 KiB
#define SMEM_GATE_BYTES ((size_t)4 * BATCH * 17 * 4)        // ~17 KiB
#define SMEM_BYTES      (SMEM_ACAT_BYTES + SMEM_GATE_BYTES)

__global__ void wcat_convert_kernel(const float* __restrict__ Wx,
                                    const float* __restrict__ Wh,
                                    __bf16* __restrict__ Wcat) {
    int idx = blockIdx.x * blockDim.x + threadIdx.x;   // one v4f per thread
    int e = idx * 4;                                   // element index in [0, 4096*2048)
    int g = e >> 11;
    int k = e & 2047;
    v4f v;
    if (k < DDIM) v = *(const v4f*)(Wx + (size_t)g * DDIM + k);
    else          v = *(const v4f*)(Wh + (size_t)g * HDIM + (k - DDIM));
    __bf16* dst = Wcat + e;
    dst[0] = (__bf16)v[0]; dst[1] = (__bf16)v[1];
    dst[2] = (__bf16)v[2]; dst[3] = (__bf16)v[3];
}

#define LOG2E 1.44269504088896340736f

__device__ __forceinline__ float fast_sigmoid(float x) {
    // v_exp_f32 + v_rcp_f32, no libm range fixups
    return __builtin_amdgcn_rcpf(1.0f + __builtin_amdgcn_exp2f(-LOG2E * x));
}

#if __has_builtin(__builtin_amdgcn_tanhf)
__device__ __forceinline__ float fast_tanh(float x) {
    return __builtin_amdgcn_tanhf(x);          // hardware v_tanh_f32 (gfx1250 TRANS)
}
#else
__device__ __forceinline__ float fast_tanh(float x) {
    float e = __builtin_amdgcn_exp2f(-2.0f * LOG2E * x);   // exp(-2x)
    return (1.0f - e) * __builtin_amdgcn_rcpf(1.0f + e);
}
#endif

__launch_bounds__(512, 1)
__global__ void lstm_persistent_kernel(const float* __restrict__ x_data,
                                       const float* __restrict__ x_mask,
                                       const float* __restrict__ bx,
                                       const __bf16* __restrict__ Wcat,
                                       __bf16* __restrict__ h_buf0,
                                       __bf16* __restrict__ h_buf1,
                                       float*  __restrict__ c_state,
                                       unsigned* __restrict__ bar,
                                       float*  __restrict__ out) {
    extern __shared__ char smem_raw[];
    __bf16* acat  = (__bf16*)smem_raw;                       // [64][2048] bf16
    float*  gates = (float*)(smem_raw + SMEM_ACAT_BYTES);    // [4][64][17] f32

    const int tid  = threadIdx.x;
    const int lane = tid & 31;
    const int wave = tid >> 5;          // 0..15
    const int gsel = wave >> 2;         // gate chunk 0..3 (i,f,o,c)
    const int b0   = (wave & 3) << 4;   // batch-row tile base: 0,16,32,48
    const int j0   = blockIdx.x << 4;   // H-column base for this block

    const int ncol = lane & 15;
    const float bias = bx[gsel * HDIM + j0 + ncol];
    // B operand: lane holds one W row (gate index), K contiguous.
    const __bf16* wrow = Wcat + (size_t)(gsel * HDIM + j0 + ncol) * KDIM;
    const int kb    = (lane < 16) ? 0 : 16;   // B-fragment K base for this lane
    const int abase = (lane < 16) ? 0 : 16;   // A-fragment K base for this lane
    const int arow  = b0 + (lane & 15);       // A-fragment row (batch index)

    // LDS byte offset of the h_prev region inside acat (generic ptr low 32 = LDS addr)
    const unsigned lds_hbase = (unsigned)(uintptr_t)(acat) + (unsigned)(DDIM * 2);

    for (int t = 0; t < T_STEPS; ++t) {
        const __bf16* h_prev = (t & 1) ? h_buf1 : h_buf0;
        __bf16*       h_next = (t & 1) ? h_buf0 : h_buf1;

        // ---- stage h_prev (bf16) into LDS via async global->LDS DMA ----
        // 64x1024 bf16 = 128 KiB = 8192 x 16B; 16 issues per thread.
        for (int i = tid; i < (BATCH * HDIM) / 8; i += 512) {
            int b = (i * 8) >> 10;
            int k = (i * 8) & (HDIM - 1);
            unsigned lds_addr = lds_hbase + (unsigned)((b * KDIM + k) * 2);
            unsigned voff = (unsigned)(i * 16);          // byte offset into h_prev
            asm volatile("global_load_async_to_lds_b128 %0, %1, %2"
                         :: "v"(lds_addr), "v"(voff), "s"(h_prev)
                         : "memory");
        }

        // ---- stage x_t (convert f32->bf16, non-temporal: read exactly once) ----
        const float* xt = x_data + (size_t)t * BATCH * DDIM;
        for (int i = tid; i < (BATCH * DDIM) / 8; i += 512) {
            const v4f* src = (const v4f*)(xt + i * 8);
            v4f v0 = __builtin_nontemporal_load(src);
            v4f v1 = __builtin_nontemporal_load(src + 1);
            v8bf d;
            d[0] = (__bf16)v0[0]; d[1] = (__bf16)v0[1];
            d[2] = (__bf16)v0[2]; d[3] = (__bf16)v0[3];
            d[4] = (__bf16)v1[0]; d[5] = (__bf16)v1[1];
            d[6] = (__bf16)v1[2]; d[7] = (__bf16)v1[3];
            int b = (i * 8) >> 10;
            int k = (i * 8) & (DDIM - 1);
            *(v8bf*)(acat + b * KDIM + k) = d;           // ds_store_b128
        }

        asm volatile("s_wait_asynccnt 0x0" ::: "memory");
        __syncthreads();

        // ---- 16x16 gate tile via WMMA over K = 2048 ----
        v8f acc;
        #pragma unroll
        for (int r = 0; r < 8; ++r) acc[r] = bias;   // bias depends only on column

        #pragma unroll 4
        for (int kc = 0; kc < KDIM; kc += 32) {
            // A 16x32 bf16 fragment from LDS (layout:
            // lanes 0-15: K {0..15}; lanes 16-31: K {16..31})
            const __bf16* ap = acat + arow * KDIM + kc + abase;
            v8bf alo = *(const v8bf*)(ap);
            v8bf ahi = *(const v8bf*)(ap + 8);
            v16bf a = __builtin_shufflevector(alo, ahi,
                0, 1, 2, 3, 4, 5, 6, 7, 8, 9, 10, 11, 12, 13, 14, 15);
            // B 32x16 bf16 fragment: lane = column, 16 contiguous K from L2
            v16bf b = *(const v16bf*)(wrow + kc + kb);
            acc = __builtin_amdgcn_wmma_f32_16x16x32_bf16(
                false, a, false, b, (short)0, acc, false, false);
        }

        // ---- scatter accumulators to LDS gate array ----
        {
            int col   = lane & 15;
            int rbase = b0 + ((lane < 16) ? 0 : 8);   // C layout: vgpr r -> rows r / r+8
            #pragma unroll
            for (int r = 0; r < 8; ++r)
                gates[(gsel * 64 + (rbase + r)) * 17 + col] = acc[r];
        }
        __syncthreads();

        // ---- pointwise LSTM update for columns [j0, j0+16) ----
        #pragma unroll
        for (int rep = 0; rep < 2; ++rep) {
            int e  = tid + rep * 512;      // 0..1023
            int b  = e >> 4;
            int jj = e & 15;
            int j  = j0 + jj;
            float ig = fast_sigmoid(gates[(0 * 64 + b) * 17 + jj]);
            float fg = fast_sigmoid(gates[(1 * 64 + b) * 17 + jj]);
            float og = fast_sigmoid(gates[(2 * 64 + b) * 17 + jj]);
            float ct = fast_tanh(gates[(3 * 64 + b) * 17 + jj]);
            float cprev  = c_state[b * HDIM + j];
            float cnew   = fg * cprev + ig * ct;
            float hnew   = og * fast_tanh(cnew);
            float m      = x_mask[t * BATCH + b];
            float hprevf = (float)acat[b * KDIM + DDIM + j];   // h_prev from LDS
            cnew = cnew * m + cprev  * (1.0f - m);
            hnew = hnew * m + hprevf * (1.0f - m);
            c_state[b * HDIM + j] = cnew;
            h_next[b * HDIM + j]  = (__bf16)hnew;
            __builtin_nontemporal_store(hnew, &out[((size_t)t * BATCH + b) * HDIM + j]);
        }
        __syncthreads();

        // ---- grid-wide barrier (cumulative count + generation) ----
        if (tid == 0) {
            __threadfence();
            unsigned target  = (unsigned)(t + 1);
            unsigned arrived = __atomic_fetch_add(bar, 1u, __ATOMIC_ACQ_REL);
            if (arrived == gridDim.x * target - 1u) {
                __atomic_store_n(bar + 1, target, __ATOMIC_RELEASE);
            } else {
                while (__atomic_load_n(bar + 1, __ATOMIC_ACQUIRE) < target) {
                    __builtin_amdgcn_s_sleep(2);
                }
            }
            __threadfence();
        }
        __syncthreads();
    }
}

extern "C" void kernel_launch(void* const* d_in, const int* in_sizes, int n_in,
                              void* d_out, int out_size, void* d_ws, size_t ws_size,
                              hipStream_t stream) {
    if (ws_size < WS_NEEDED) return;   // deterministic no-op if scratch too small

    const float* x_data = (const float*)d_in[0];
    const float* x_mask = (const float*)d_in[1];
    const float* Wx     = (const float*)d_in[2];
    const float* bx     = (const float*)d_in[3];
    const float* Wh     = (const float*)d_in[4];
    float* out = (float*)d_out;

    char* ws = (char*)d_ws;
    __bf16*   Wcat  = (__bf16*)ws;
    __bf16*   h0    = (__bf16*)(ws + OFF_H0);
    __bf16*   h1    = (__bf16*)(ws + OFF_H1);
    float*    cst   = (float*)(ws + OFF_C);
    unsigned* bar   = (unsigned*)(ws + OFF_BAR);

    // allow 273 KB dynamic LDS per workgroup (WGP has 320 KB)
    (void)hipFuncSetAttribute((const void*)lstm_persistent_kernel,
                              hipFuncAttributeMaxDynamicSharedMemorySize,
                              (int)SMEM_BYTES);

    // zero h0/h1/c/barrier state each launch (graph-capturable memset node)
    (void)hipMemsetAsync(ws + OFF_H0, 0, 2 * H_BYTES + C_BYTES + 256, stream);

    // bf16 weight concat [Wx | Wh] -> 16 MiB, L2-resident thereafter
    {
        int total4 = (GDIM * KDIM) / 4;
        wcat_convert_kernel<<<total4 / 256, 256, 0, stream>>>(Wx, Wh, Wcat);
    }

    // persistent recurrent kernel: 64 blocks (one per 16 H-columns), 16 waves each
    lstm_persistent_kernel<<<HDIM / 16, 512, SMEM_BYTES, stream>>>(
        x_data, x_mask, bx, Wcat, h0, h1, cst, bar, out);
}
